// GATLayer_35003983462712
// MI455X (gfx1250) — compile-verified, matrix operand order checked by reading, not run on hardware
//
#include <hip/hip_runtime.h>
#include <hip/hip_bf16.h>
#include <math.h>

typedef __attribute__((ext_vector_type(16))) _Float16 v16h;
typedef __attribute__((ext_vector_type(8)))  _Float16 v8h;
typedef __attribute__((ext_vector_type(8)))  float    v8f;

union F16x16 { v16h v; v8h h[2]; };

#define N_NODES   50000
#define IN_DIM    256
#define OUT_DIM   128
#define NUM_HEADS 2
#define N_COLS    (OUT_DIM * NUM_HEADS)   // 256
#define N_EDGES   800000

// ---------------- GEMM tiling ----------------
#define BM 128
#define BN 128
#define NT 8                               // 16-col tiles per wave
#define M_BLOCKS ((N_NODES + BM - 1) / BM) // 391
#define M_PAD    (M_BLOCKS * BM)           // 50048 rows in padded f16 copy of x

// ---------- ordered-uint encoding for float atomic max ----------
__device__ __forceinline__ unsigned f2ord(float f) {
    unsigned u = __float_as_uint(f);
    return (u & 0x80000000u) ? ~u : (u | 0x80000000u);
}
__device__ __forceinline__ float ord2f(unsigned u) {
    return (u & 0x80000000u) ? __uint_as_float(u & 0x7FFFFFFFu)
                             : __uint_as_float(~u);
}
#define ORD_NEG_INF 0x007FFFFFu   // f2ord(-inf)

// ============ prep: x -> f16 (zero-padded rows), W -> f16 transposed ============
__global__ void gat_prep_xh(const float* __restrict__ x, _Float16* __restrict__ xh)
{
    int i = blockIdx.x * blockDim.x + threadIdx.x;
    if (i >= M_PAD * IN_DIM) return;
    int row = i / IN_DIM;
    xh[i] = (row < N_NODES) ? (_Float16)x[i] : (_Float16)0.0f;
}

__global__ void gat_prep_wt(const float* __restrict__ W, _Float16* __restrict__ wt)
{
    int i = blockIdx.x * blockDim.x + threadIdx.x;   // n*IN_DIM + k
    if (i >= N_COLS * IN_DIM) return;
    int n = i / IN_DIM, k = i % IN_DIM;
    wt[i] = (_Float16)W[(size_t)k * N_COLS + n];
}

// ================= Phase 1: Wh = x @ W + b (WMMA f16 -> f32) =================
// Direct-from-L2 GEMM: both operands are L2-resident; every fragment is two
// global_load_b128. No LDS, no barriers.
__global__ __launch_bounds__(256)
void gat_gemm_wmma(const _Float16* __restrict__ xh, const _Float16* __restrict__ wt,
                   const float* __restrict__ bias, float* __restrict__ Wh)
{
    const int lane = threadIdx.x & 31;
    const int wave = threadIdx.x >> 5;
    const int m0   = blockIdx.x * BM;
    const int n0   = blockIdx.y * BN;

    // A: lane l holds row (wave*16 + l%16); lanes 0-15 K={0..7,16..23}, 16-31 K={8..15,24..31}
    const int arow = m0 + wave * 16 + (lane & 15);
    const int kb   = (lane < 16) ? 0 : 8;
    const _Float16* ap = xh + (size_t)arow * IN_DIM + kb;
    // B: lane l holds col (n0 + t*16 + l%16), same K pattern, from N-major wt
    const _Float16* bp = wt + (size_t)(n0 + (lane & 15)) * IN_DIM + kb;

    v8f acc[NT] = {};
#pragma unroll
    for (int k0 = 0; k0 < IN_DIM; k0 += 32) {
        F16x16 af;
        af.h[0] = *(const v8h*)(ap + k0);
        af.h[1] = *(const v8h*)(ap + k0 + 16);
#pragma unroll
        for (int t = 0; t < NT; ++t) {
            F16x16 bf;
            const _Float16* bt = bp + (size_t)t * 16 * IN_DIM + k0;
            bf.h[0] = *(const v8h*)(bt);
            bf.h[1] = *(const v8h*)(bt + 16);
            acc[t] = __builtin_amdgcn_wmma_f32_16x16x32_f16(
                false, af.v, false, bf.v, (short)0, acc[t], false, false);
        }
    }

    // C/D layout: lane l, vgpr v -> row = v + (l>=16 ? 8 : 0), col = l%16
    const int rbase = m0 + wave * 16 + ((lane < 16) ? 0 : 8);
    const int cl    = lane & 15;
#pragma unroll
    for (int t = 0; t < NT; ++t) {
        int col = n0 + t * 16 + cl;
        float bv = bias[col];
#pragma unroll
        for (int v = 0; v < 8; ++v) {
            int row = rbase + v;
            if (row < N_NODES)
                Wh[(size_t)row * N_COLS + col] = acc[t][v] + bv;
        }
    }
}

// ================= Phase 2 kernels (edge softmax + aggregate) =================
__global__ void gat_init(float* __restrict__ out, unsigned* __restrict__ mord,
                         float* __restrict__ denom, float* __restrict__ deg)
{
    int i = blockIdx.x * blockDim.x + threadIdx.x;
    if (i < N_NODES * N_COLS) out[i] = 0.0f;
    if (i < N_NODES * NUM_HEADS) { mord[i] = ORD_NEG_INF; denom[i] = 0.0f; }
    if (i < N_NODES) deg[i] = 0.0f;
}

__global__ void gat_st(const float* __restrict__ Wh, const float* __restrict__ a,
                       float* __restrict__ s, float* __restrict__ t)
{
    int i = blockIdx.x * blockDim.x + threadIdx.x;   // node*H + h
    if (i >= N_NODES * NUM_HEADS) return;
    int node = i / NUM_HEADS, h = i % NUM_HEADS;
    const float* wh = Wh + (size_t)node * N_COLS + h * OUT_DIM;
    float ss = 0.f, tt = 0.f;
#pragma unroll 8
    for (int d = 0; d < OUT_DIM; ++d) {
        float v = wh[d];
        ss += v * a[d];
        tt += v * a[OUT_DIM + d];
    }
    s[i] = ss; t[i] = tt;
}

__global__ void gat_edge_max(const int* __restrict__ src, const int* __restrict__ dst,
                             const float* __restrict__ s, const float* __restrict__ t,
                             float* __restrict__ e, unsigned* __restrict__ mord,
                             float* __restrict__ deg)
{
    int i = blockIdx.x * blockDim.x + threadIdx.x;   // edge*H + h
    if (i >= N_EDGES * NUM_HEADS) return;
    int ei = i >> 1, h = i & 1;
    int u = src[ei], v = dst[ei];
    float val = s[u * NUM_HEADS + h] + t[v * NUM_HEADS + h];
    val = (val > 0.f) ? val : 0.2f * val;            // leaky_relu(0.2)
    e[i] = val;
    atomicMax(&mord[u * NUM_HEADS + h], f2ord(val));
    if (h == 0) atomicAdd(&deg[u], 1.0f);
}

__global__ void gat_edge_exp(const int* __restrict__ src,
                             float* __restrict__ e,          // in: logits, out: exp
                             const unsigned* __restrict__ mord,
                             float* __restrict__ denom)
{
    int i = blockIdx.x * blockDim.x + threadIdx.x;
    if (i >= N_EDGES * NUM_HEADS) return;
    int ei = i >> 1, h = i & 1;
    int u = src[ei];
    float ex = __expf(e[i] - ord2f(mord[u * NUM_HEADS + h]));
    e[i] = ex;
    atomicAdd(&denom[u * NUM_HEADS + h], ex);
}

// 32 lanes per (edge, head); each lane scatters 4 output dims
__global__ void gat_edge_scatter(const int* __restrict__ src, const int* __restrict__ dst,
                                 const float* __restrict__ ex, const float* __restrict__ denom,
                                 const float* __restrict__ Wh, float* __restrict__ out)
{
    int gid  = blockIdx.x * blockDim.x + threadIdx.x;
    int lane = gid & 31;
    int grp  = gid >> 5;                    // edge*H + h
    if (grp >= N_EDGES * NUM_HEADS) return;
    int ei = grp >> 1, h = grp & 1;
    int u = src[ei], v = dst[ei];
    float alpha = ex[grp] / denom[u * NUM_HEADS + h];
    const float* whr = Wh  + (size_t)v * N_COLS + h * OUT_DIM + lane * 4;
    float*      outr = out + (size_t)u * N_COLS + h * OUT_DIM + lane * 4;
#pragma unroll
    for (int j = 0; j < 4; ++j)
        atomicAdd(&outr[j], alpha * whr[j]);
}

__global__ void gat_finalize(const float* __restrict__ deg, const float* __restrict__ Wh,
                             float* __restrict__ out)
{
    int i = blockIdx.x * blockDim.x + threadIdx.x;
    if (i >= N_NODES * N_COLS) return;
    int node = i / N_COLS;
    if (deg[node] <= 0.f) out[i] = Wh[i];
}

// ================================ launcher ================================
extern "C" void kernel_launch(void* const* d_in, const int* in_sizes, int n_in,
                              void* d_out, int out_size, void* d_ws, size_t ws_size,
                              hipStream_t stream)
{
    const float* x   = (const float*)d_in[0];
    const int*   ei  = (const int*)d_in[1];          // (2, N_EDGES): [src | dst]
    const float* W   = (const float*)d_in[2];
    const float* b   = (const float*)d_in[3];
    const float* a   = (const float*)d_in[4];
    float*       out = (float*)d_out;

    const int* src = ei;
    const int* dst = ei + N_EDGES;

    // workspace carve-up
    char* ws = (char*)d_ws;
    size_t off = 0;
    float*     Wh    = (float*)(ws + off);     off += (size_t)N_NODES * N_COLS * 4;      // 51.2 MB
    _Float16*  xh    = (_Float16*)(ws + off);  off += (size_t)M_PAD * IN_DIM * 2;        // 25.6 MB
    _Float16*  wt    = (_Float16*)(ws + off);  off += (size_t)N_COLS * IN_DIM * 2;       // 128 KB
    float*     s     = (float*)(ws + off);     off += (size_t)N_NODES * NUM_HEADS * 4;
    float*     t     = (float*)(ws + off);     off += (size_t)N_NODES * NUM_HEADS * 4;
    unsigned*  mord  = (unsigned*)(ws + off);  off += (size_t)N_NODES * NUM_HEADS * 4;
    float*     denom = (float*)(ws + off);     off += (size_t)N_NODES * NUM_HEADS * 4;
    float*     deg   = (float*)(ws + off);     off += (size_t)N_NODES * 4;
    float*     e     = (float*)(ws + off);     off += (size_t)N_EDGES * NUM_HEADS * 4;   // 6.4 MB
    (void)off; (void)ws_size; (void)in_sizes; (void)n_in; (void)out_size;

    // 0) init accumulators / out
    {
        int n = N_NODES * N_COLS;
        gat_init<<<(n + 255) / 256, 256, 0, stream>>>(out, mord, denom, deg);
    }
    // 0b) precision prep: xh = f16(x) padded; wt = f16(W^T)
    {
        int n = M_PAD * IN_DIM;
        gat_prep_xh<<<(n + 255) / 256, 256, 0, stream>>>(x, xh);
        int m = N_COLS * IN_DIM;
        gat_prep_wt<<<(m + 255) / 256, 256, 0, stream>>>(W, wt);
    }
    // 1) Wh = x @ W + b   (WMMA, direct-from-L2)
    {
        dim3 grid(M_BLOCKS, N_COLS / BN);
        gat_gemm_wmma<<<grid, 256, 0, stream>>>(xh, wt, b, Wh);
    }
    // 2) per-node attention logit halves s, t
    {
        int n = N_NODES * NUM_HEADS;
        gat_st<<<(n + 255) / 256, 256, 0, stream>>>(Wh, a, s, t);
    }
    // 3) edge logits + segment max + degree
    {
        int n = N_EDGES * NUM_HEADS;
        gat_edge_max<<<(n + 255) / 256, 256, 0, stream>>>(src, dst, s, t, e, mord, deg);
    }
    // 4) exp + segment sum
    {
        int n = N_EDGES * NUM_HEADS;
        gat_edge_exp<<<(n + 255) / 256, 256, 0, stream>>>(src, e, mord, denom);
    }
    // 5) alpha-weighted scatter into out
    {
        long long n = (long long)N_EDGES * NUM_HEADS * 32;
        gat_edge_scatter<<<(unsigned)((n + 255) / 256), 256, 0, stream>>>(src, dst, e, denom, Wh, out);
    }
    // 6) isolated-node fallback
    {
        int n = N_NODES * N_COLS;
        gat_finalize<<<(n + 255) / 256, 256, 0, stream>>>(deg, Wh, out);
    }
}
